// BinaryFactoredLinear_62947040690668
// MI455X (gfx1250) — matrix operand order repaired and checked by reading, not compiled, not run on hardware
//
#include <hip/hip_runtime.h>

// Problem constants (fixed by the reference):
//   x[B=4, S=4096, D_IN=4096]  -> M_TOT = 16384 rows
//   U_latent[D_OUT=4096, R=128], V_latent[D_IN=4096, R=128]
#define M_TOT  16384
#define D_IN   4096
#define D_OUT  4096
#define R_DIM  128

typedef __bf16        bf16x16 __attribute__((ext_vector_type(16)));
typedef __bf16        bf16x2  __attribute__((ext_vector_type(2)));
typedef float         floatx8 __attribute__((ext_vector_type(8)));
typedef float         floatx4 __attribute__((ext_vector_type(4)));
typedef unsigned int  uintx4  __attribute__((ext_vector_type(4)));

union Frag16 {           // 16 bf16 values = 8 VGPRs, loadable as two 16B chunks
    uintx4  u[2];
    bf16x16 v;
};

__device__ __forceinline__ unsigned int pack2bf(float lo, float hi) {
#if __has_builtin(__builtin_amdgcn_cvt_pk_bf16_f32)
    bf16x2 p = __builtin_amdgcn_cvt_pk_bf16_f32(lo, hi);
#else
    // native fptrunc: backend can pair these into v_cvt_pk_bf16_f32
    bf16x2 p;
    p.x = (__bf16)lo;
    p.y = (__bf16)hi;
#endif
    return __builtin_bit_cast(unsigned int, p);
}

__device__ __forceinline__ unsigned short f2bf(float f) {
    __bf16 b = (__bf16)f;
    return __builtin_bit_cast(unsigned short, b);
}

__device__ __forceinline__ unsigned short signbf(float f) {
    // sign(x) in {-1, 0, +1}, exact in bf16
    return f > 0.0f ? 0x3F80u : (f < 0.0f ? 0xBF80u : 0u);
}

// ---------------------------------------------------------------------------
// Kernel 0a: Vt[r][k] = sign(V_latent[k][r])  (bf16, transposed so stage-1 B
// fragments are two contiguous 8-element runs along k)
// ---------------------------------------------------------------------------
__global__ __launch_bounds__(256) void sign_transpose_V(
    const float* __restrict__ Vl, unsigned short* __restrict__ Vt)
{
    const int e = blockIdx.x * 256 + threadIdx.x;   // e = k*128 + r
    const int k = e >> 7;
    const int r = e & 127;
    Vt[(size_t)r * D_IN + k] = signbf(Vl[e]);
}

// ---------------------------------------------------------------------------
// Kernel 0b: Us[d][r] = sign(U_latent[d][r])  (bf16, already the layout the
// stage-2 B fragments want: contiguous along r)
// ---------------------------------------------------------------------------
__global__ __launch_bounds__(256) void sign_U(
    const float* __restrict__ Ul, unsigned short* __restrict__ Us)
{
    const int e = blockIdx.x * 256 + threadIdx.x;
    Us[e] = signbf(Ul[e]);
}

// ---------------------------------------------------------------------------
// Kernel 1: Z1[M_TOT, 128] = bf16( (x * s2) @ sign(V) )
//   WG = 256 threads = 8 waves. Tile = 64 rows x R=128 (256 WGs for occupancy).
//   Waves: 4 M-tiles x 2 N-halves (4 N-tiles of 16 each).
//   Double-buffered LDS: global loads for step k+1 issue before the WMMAs of
//   step k; a single workgroup barrier per iteration.
// ---------------------------------------------------------------------------
#define K1_BM 64
#define K1_KC 32
#define LSTR  40      // LDS row stride in bf16 elements (80B: 16B-aligned, conflict-free)

__global__ __launch_bounds__(256) void stage1_gemm(
    const float* __restrict__ x, const float* __restrict__ s2,
    const unsigned short* __restrict__ Vt, unsigned short* __restrict__ Z1)
{
    __shared__ unsigned short xs[2][K1_BM * LSTR];   // bf16(x*s2) tile [64 m][32 k]
    __shared__ unsigned short vs[2][R_DIM * LSTR];   // Vt tile       [128 r][32 k]

    const int tid   = threadIdx.x;
    const int wave  = tid >> 5;
    const int lane  = tid & 31;
    const int half  = lane >> 4;     // 0: lanes 0-15, 1: lanes 16-31
    const int nlane = lane & 15;
    const int mw    = wave & 3;      // wave's M-tile (0..3)
    const int ng    = wave >> 2;     // wave's N-half (0..1), 4 N-tiles each
    const int m0    = blockIdx.x * K1_BM;

    // staging coords: x -> (row 0..63, 8-float quarter), Vt -> (row 0..127, 16-half)
    const int xr = tid >> 2;
    const int xq = (tid & 3) * 8;
    const int vr = tid >> 1;
    const int vh = (tid & 1) * 16;

    const float*          xrow = x  + (size_t)(m0 + xr) * D_IN + xq;
    const unsigned short* vrow = Vt + (size_t)vr        * D_IN + vh;

    floatx8 acc[4];
    #pragma unroll
    for (int t = 0; t < 4; ++t) acc[t] = {};

    // ---- prologue: stage tile kk=0 into buffer 0 ----
    {
        floatx4 x0 = *(const floatx4*)(xrow + 0);
        floatx4 x1 = *(const floatx4*)(xrow + 4);
        floatx4 s0 = *(const floatx4*)(s2 + xq + 0);
        floatx4 s1v = *(const floatx4*)(s2 + xq + 4);
        uintx4 p;
        p.x = pack2bf(x0.x * s0.x, x0.y * s0.y);
        p.y = pack2bf(x0.z * s0.z, x0.w * s0.w);
        p.z = pack2bf(x1.x * s1v.x, x1.y * s1v.y);
        p.w = pack2bf(x1.z * s1v.z, x1.w * s1v.w);
        *(uintx4*)&xs[0][xr * LSTR + xq] = p;

        uintx4 v0 = *(const uintx4*)(vrow + 0);
        uintx4 v1 = *(const uintx4*)(vrow + 8);
        *(uintx4*)&vs[0][vr * LSTR + vh]     = v0;
        *(uintx4*)&vs[0][vr * LSTR + vh + 8] = v1;
    }
    __syncthreads();

    int cur = 0;
    for (int kk = 0; kk < D_IN; kk += K1_KC) {
        const bool has_next = (kk + K1_KC) < D_IN;

        // ---- issue next tile's global loads early (overlap with WMMA) ----
        floatx4 nx0, nx1, ns0, ns1;
        uintx4  nv0, nv1;
        if (has_next) {
            nx0 = *(const floatx4*)(xrow + kk + K1_KC + 0);
            nx1 = *(const floatx4*)(xrow + kk + K1_KC + 4);
            ns0 = *(const floatx4*)(s2 + kk + K1_KC + xq + 0);
            ns1 = *(const floatx4*)(s2 + kk + K1_KC + xq + 4);
            nv0 = *(const uintx4*)(vrow + kk + K1_KC);
            nv1 = *(const uintx4*)(vrow + kk + K1_KC + 8);
            __builtin_prefetch(xrow + kk + 2 * K1_KC, 0, 0);   // global_prefetch_b8
        }

        // ---- A fragment (16x32 bf16): lane half -> K runs {h8..+7},{16+h8..+7}
        Frag16 a;
        const unsigned short* abase = &xs[cur][(16 * mw + nlane) * LSTR + 8 * half];
        a.u[0] = *(const uintx4*)(abase);        // ds_load_b128
        a.u[1] = *(const uintx4*)(abase + 16);

        // ---- 4 N-tiles, B fragment (32x16 bf16): lane half -> K runs {h8..+7},{16+h8..+7}
        #pragma unroll
        for (int t = 0; t < 4; ++t) {
            Frag16 b;
            const unsigned short* bbase =
                &vs[cur][(64 * ng + 16 * t + nlane) * LSTR + 8 * half];
            b.u[0] = *(const uintx4*)(bbase);
            b.u[1] = *(const uintx4*)(bbase + 16);
            acc[t] = __builtin_amdgcn_wmma_f32_16x16x32_bf16(
                false, a.v, false, b.v, (short)0, acc[t], false, false);
        }

        // ---- convert + store next tile into the other buffer ----
        if (has_next) {
            uintx4 p;
            p.x = pack2bf(nx0.x * ns0.x, nx0.y * ns0.y);
            p.y = pack2bf(nx0.z * ns0.z, nx0.w * ns0.w);
            p.z = pack2bf(nx1.x * ns1.x, nx1.y * ns1.y);
            p.w = pack2bf(nx1.z * ns1.z, nx1.w * ns1.w);
            *(uintx4*)&xs[cur ^ 1][xr * LSTR + xq]       = p;
            *(uintx4*)&vs[cur ^ 1][vr * LSTR + vh]       = nv0;
            *(uintx4*)&vs[cur ^ 1][vr * LSTR + vh + 8]   = nv1;
        }
        __syncthreads();
        cur ^= 1;
    }

    // ---- epilogue: Z1 bf16. C layout: lane -> col nlane, rows 8*half + v
    #pragma unroll
    for (int t = 0; t < 4; ++t) {
        const int n = 64 * ng + 16 * t + nlane;
        unsigned short* zp = Z1 + (size_t)(m0 + 16 * mw + 8 * half) * R_DIM + n;
        #pragma unroll
        for (int v = 0; v < 8; ++v)
            zp[(size_t)v * R_DIM] = f2bf(acc[t][v]);
    }
}

// ---------------------------------------------------------------------------
// Kernel 2: out[M_TOT, D_OUT] = Z1 @ sign(U)^T * s1 + bias
//   K=128 held entirely in registers (4 A fragments per wave). The 128x128
//   B tile (32KB) is staged in LDS once per WG so the 8 waves don't each
//   re-pull it from cache. Output store (268MB) dominates HBM traffic.
// ---------------------------------------------------------------------------
#define USTR 136   // LDS row stride (272B: 16B-aligned, conflict-free b128 reads)

__global__ __launch_bounds__(256) void stage2_gemm(
    const unsigned short* __restrict__ Z1, const unsigned short* __restrict__ Us,
    const float* __restrict__ s1, const float* __restrict__ bias,
    float* __restrict__ out)
{
    __shared__ unsigned short us[128 * USTR];   // sign(U) tile [128 n][128 k]

    const int tid   = threadIdx.x;
    const int wave  = tid >> 5;
    const int lane  = tid & 31;
    const int half  = lane >> 4;
    const int nlane = lane & 15;
    const int m0 = blockIdx.y * 128 + 16 * wave;
    const int n0 = blockIdx.x * 128;

    // ---- stage B tile: thread -> (row 0..127, 64-col half), 8x b128 each ----
    {
        const int r  = tid >> 1;
        const int c0 = (tid & 1) * 64;
        const unsigned short* ug = Us + (size_t)(n0 + r) * R_DIM + c0;
        #pragma unroll
        for (int i = 0; i < 8; ++i) {
            uintx4 q = *(const uintx4*)(ug + 8 * i);
            *(uintx4*)&us[r * USTR + c0 + 8 * i] = q;
        }
    }

    // ---- A fragments for this wave's 16-row tile (Z1 row = 128 bf16) ----
    Frag16 a[4];
    const unsigned short* zrow = Z1 + (size_t)(m0 + nlane) * R_DIM;
    #pragma unroll
    for (int ks = 0; ks < 4; ++ks) {
        a[ks].u[0] = *(const uintx4*)(zrow + 32 * ks + 8 * half);
        a[ks].u[1] = *(const uintx4*)(zrow + 32 * ks + 16 + 8 * half);
    }

    __syncthreads();

    #pragma unroll
    for (int t = 0; t < 8; ++t) {
        const int nl = 16 * t + nlane;           // local col in tile
        const int n  = n0 + nl;                  // global output col
        const unsigned short* urow = &us[nl * USTR];
        floatx8 acc = {};
        #pragma unroll
        for (int ks = 0; ks < 4; ++ks) {
            Frag16 b;
            b.u[0] = *(const uintx4*)(urow + 32 * ks + 8 * half);
            b.u[1] = *(const uintx4*)(urow + 32 * ks + 8 * half + 16);
            acc = __builtin_amdgcn_wmma_f32_16x16x32_bf16(
                false, a[ks].v, false, b.v, (short)0, acc, false, false);
        }
        const float sc = s1[n];
        const float bb = bias[n];
        float* op = out + (size_t)(m0 + 8 * half) * D_OUT + n;
        #pragma unroll
        for (int v = 0; v < 8; ++v)
            op[(size_t)v * D_OUT] = fmaf(acc[v], sc, bb);
    }
}

// ---------------------------------------------------------------------------
extern "C" void kernel_launch(void* const* d_in, const int* in_sizes, int n_in,
                              void* d_out, int out_size, void* d_ws, size_t ws_size,
                              hipStream_t stream) {
    const float* x    = (const float*)d_in[0];
    const float* Ul   = (const float*)d_in[1];
    const float* Vl   = (const float*)d_in[2];
    const float* s1   = (const float*)d_in[3];
    const float* s2   = (const float*)d_in[4];
    const float* bias = (const float*)d_in[5];
    float* out = (float*)d_out;

    // Workspace layout (bf16): Vt[128,4096] | Us[4096,128] | Z1[16384,128] = 6 MB
    unsigned short* Vt = (unsigned short*)d_ws;
    unsigned short* Us = Vt + (size_t)R_DIM * D_IN;
    unsigned short* Z1 = Us + (size_t)D_OUT * R_DIM;

    sign_transpose_V<<<(D_IN * R_DIM) / 256, 256, 0, stream>>>(Vl, Vt);
    sign_U<<<(D_OUT * R_DIM) / 256, 256, 0, stream>>>(Ul, Us);
    stage1_gemm<<<M_TOT / K1_BM, 256, 0, stream>>>(x, s2, Vt, Z1);
    stage2_gemm<<<dim3(D_OUT / 128, M_TOT / 128), 256, 0, stream>>>(Z1, Us, s1, bias, out);
}
